// Segmentation_16587163697763
// MI455X (gfx1250) — matrix operand order repaired and checked by reading, not compiled, not run on hardware
//
#include <hip/hip_runtime.h>
#include <hip/hip_bf16.h>

typedef __attribute__((ext_vector_type(16))) _Float16 v16h;
typedef __attribute__((ext_vector_type(8)))  float    v8f;

// ---------------------------------------------------------------------------
// Weight conversion: f32 coeff[2,O,I,32] -> f16 fragment-swizzled buffer.
// Consumed layout (B-fragment for v_wmma_f32_16x16x32_f16, wave32):
//   frag F = kt*(O/16) + nto ; element = Wf[(F*32 + lane)*16 + j]
//   col  n = nto*16 + (lane&15)
//   K      = kt*32 + (lane>>4)*16 + j       (K = i*64 + trig*32 + g)
// ---------------------------------------------------------------------------
__global__ __launch_bounds__(256)
void kan_conv_weights(const float* __restrict__ src, _Float16* __restrict__ dst,
                      int I, int O) {
  long total = (long)I * 64 * O;
  long idx = (long)blockIdx.x * 256 + threadIdx.x;
  if (idx >= total) return;
  int j    = (int)(idx & 15);
  int lane = (int)((idx >> 4) & 31);
  long rest = idx >> 9;
  int nt_o = O >> 4;
  int nto  = (int)(rest % nt_o);
  int kt   = (int)(rest / nt_o);
  int n = nto * 16 + (lane & 15);
  int K = kt * 32 + (lane >> 4) * 16 + j;
  int i = K >> 6, rem = K & 63, t = rem >> 5, g = rem & 31;
  dst[idx] = (_Float16)src[(((long)t * O + n) * I + i) * 32 + g];
}

// ---------------------------------------------------------------------------
// LayerNorm over rows of length L (biased var, eps=1e-5). One wave per row.
// ---------------------------------------------------------------------------
__global__ __launch_bounds__(256)
void ln_rows(const float* __restrict__ in, float* __restrict__ out, int L) {
  int lane = threadIdx.x & 31;
  long row = (long)blockIdx.x * 8 + (threadIdx.x >> 5);
  const float* p = in + row * L;
  float s = 0.f, ss = 0.f;
  for (int i = lane; i < L; i += 32) { float v = p[i]; s += v; ss += v * v; }
  for (int off = 16; off; off >>= 1) {
    s  += __shfl_xor(s,  off, 32);
    ss += __shfl_xor(ss, off, 32);
  }
  float mean = s / (float)L;
  float var  = ss / (float)L - mean * mean;
  float inv  = rsqrtf(var + 1e-5f);
  float* q = out + row * L;
  for (int i = lane; i < L; i += 32) q[i] = (p[i] - mean) * inv;
}

// ---------------------------------------------------------------------------
// Fused Fourier-KAN GEMM: y[r,o] = sum_{i,g} cos/sin(x[r,i]*(g+1)) * W + b[o]
// Block = 128 threads = 4 waves. 16-row tile per block; each wave owns NT
// 16-wide output tiles (N chunk = NT*64 per block, blockIdx.y chunks).
// K-loop: 2 input elements per step = 128 K values = 4 WMMA K=32 tiles.
// phi written to LDS in exact 16x32 f16 A-fragment order.
// Generic strides: in addr  = (r/iRS)*iS0 + (r%iRS)*iS1 + i*iS2
//                  out addr = (r/oRS)*oOB + (r%oRS)*oQS + o*oOS
// ---------------------------------------------------------------------------
template <int NT>
__global__ __launch_bounds__(128)
void kan_wmma(const float* __restrict__ xin, const _Float16* __restrict__ Wf,
              const float* __restrict__ bias, float* __restrict__ out,
              int I, int O,
              int iRS, int iS0, int iS1, int iS2,
              int oRS, int oOB, int oQS, int oOS) {
  __shared__ __attribute__((aligned(32))) _Float16 phi[4 * 32 * 16]; // 4KB
  const int tid  = threadIdx.x;
  const int lane = tid & 31;
  const int w    = tid >> 5;
  const int r_base = blockIdx.x * 16;
  const int n_base = blockIdx.y * (NT * 64);
  const int nt_o   = O >> 4;

  v8f acc[NT];
  const v8f vzero = {0.f, 0.f, 0.f, 0.f, 0.f, 0.f, 0.f, 0.f};
  for (int t = 0; t < NT; ++t) acc[t] = vzero;

  // trig-producer mapping: 16 rows x 2 elems x (4 groups of 8 freqs)
  const int m   = tid & 15;       // row in tile
  const int sub = tid >> 4;       // 0..7
  const int il  = sub >> 2;       // input element 0/1
  const int gq  = sub & 3;        // frequency quarter
  const int gbase   = gq * 8;
  const int lane_hi = gq >> 1;
  const int jb      = (gq & 1) ? 8 : 0;
  const int r = r_base + m;
  const long in_row = (long)(r / iRS) * iS0 + (long)(r % iRS) * iS1;
  const int base_c = (il * 2 + 0) * 512 + (lane_hi * 16 + m) * 16 + jb;
  const int base_s = (il * 2 + 1) * 512 + (lane_hi * 16 + m) * 16 + jb;

  for (int i0 = 0; i0 < I; i0 += 2) {
    __syncthreads();
    // ---- expansion: cos/sin features into LDS (A-fragment order) ----
    float xv = xin[in_row + (long)(i0 + il) * iS2];
    #pragma unroll
    for (int g = 0; g < 8; ++g) {
      float sv, cv;
      __sincosf(xv * (float)(gbase + g + 1), &sv, &cv);
      phi[base_c + g] = (_Float16)cv;
      phi[base_s + g] = (_Float16)sv;
    }
    __syncthreads();

    // ---- WMMA over the 4 K-tiles of this chunk ----
    const long frag_base = (long)(i0 >> 1) * 4 * nt_o;
    if (i0 + 2 < I) // prefetch next weight chunk (global_prefetch path)
      __builtin_prefetch(&Wf[((frag_base + 4 * nt_o) * 32 + lane) * 16], 0, 1);
    #pragma unroll
    for (int kt = 0; kt < 4; ++kt) {
      v16h a = *(const v16h*)(phi + kt * 512 + lane * 16);
      #pragma unroll
      for (int t = 0; t < NT; ++t) {
        int nto = (n_base >> 4) + w * NT + t;
        v16h b = *(const v16h*)(Wf + ((frag_base + (long)kt * nt_o + nto) * 32
                                      + lane) * 16);
        acc[t] = __builtin_amdgcn_wmma_f32_16x16x32_f16(
            false, a, false, b, (short)0, acc[t], false, false);
      }
    }
  }

  // ---- epilogue: 16x16 f32 C layout -> strided stores + bias ----
  const int nloc = lane & 15;
  const int mh   = (lane >> 4) * 8;
  for (int t = 0; t < NT; ++t) {
    int o = n_base + (w * NT + t) * 16 + nloc;
    float bv = bias[o];
    #pragma unroll
    for (int v = 0; v < 8; ++v) {
      int rr = r_base + mh + v;
      long addr = (long)(rr / oRS) * oOB + (long)(rr % oRS) * oQS
                + (long)o * oOS;
      out[addr] = acc[t][v] + bv;
    }
  }
}

// ---------------------------------------------------------------------------
extern "C" void kernel_launch(void* const* d_in, const int* in_sizes, int n_in,
                              void* d_out, int out_size, void* d_ws,
                              size_t ws_size, hipStream_t stream) {
  const float* x      = (const float*)d_in[0];
  const float* w_pre  = (const float*)d_in[1];
  const float* b_pre  = (const float*)d_in[2];
  const float* w_amp  = (const float*)d_in[3];
  const float* b_amp  = (const float*)d_in[4];
  const float* w_mix0 = (const float*)d_in[5];
  const float* b_mix0 = (const float*)d_in[6];
  const float* w_mix1 = (const float*)d_in[7];
  const float* b_mix1 = (const float*)d_in[8];
  const float* w_mix2 = (const float*)d_in[9];
  const float* b_mix2 = (const float*)d_in[10];

  char* p = (char*)d_ws;
  auto carve = [&](size_t bytes) -> char* {
    char* q = p; p += (bytes + 255) & ~(size_t)255; return q;
  };
  _Float16* wbPre  = (_Float16*)carve((size_t)196 * 64 * 64 * 2);
  _Float16* wbAmp0 = (_Float16*)carve((size_t)256 * 64 * 512 * 2);
  _Float16* wbAmp1 = (_Float16*)carve((size_t)256 * 64 * 512 * 2);
  _Float16* wbAmp2 = (_Float16*)carve((size_t)256 * 64 * 512 * 2);
  _Float16* wbMix0 = (_Float16*)carve((size_t)64  * 64 * 64  * 2);
  _Float16* wbMix1 = (_Float16*)carve((size_t)128 * 64 * 128 * 2);
  _Float16* wbMix2 = (_Float16*)carve((size_t)256 * 64 * 256 * 2);
  float* hA  = (float*)carve((size_t)524288 * 4);
  float* hB  = (float*)carve((size_t)524288 * 4);
  float* hn  = (float*)carve((size_t)262144 * 4);
  float* ab  = (float*)carve((size_t)524288 * 4);
  float* an  = (float*)carve((size_t)524288 * 4);

  auto conv = [&](const float* src, _Float16* dst, int I, int O) {
    long total = (long)I * 64 * O;
    kan_conv_weights<<<(int)((total + 255) / 256), 256, 0, stream>>>(src, dst, I, O);
  };
  conv(w_pre, wbPre, 196, 64);
  const size_t ampStride = (size_t)2 * 512 * 256 * 32;
  conv(w_amp + 0 * ampStride, wbAmp0, 256, 512);
  conv(w_amp + 1 * ampStride, wbAmp1, 256, 512);
  conv(w_amp + 2 * ampStride, wbAmp2, 256, 512);
  conv(w_mix0, wbMix0, 64, 64);
  conv(w_mix1, wbMix1, 128, 128);
  conv(w_mix2, wbMix2, 256, 256);

  auto kan = [&](const float* in, const _Float16* Wf, const float* bias,
                 float* out, int R, int I, int O,
                 int iRS, int iS0, int iS1, int iS2,
                 int oRS, int oOB, int oQS, int oOS) {
    int NT = O / 64; if (NT > 4) NT = 4; if (NT < 1) NT = 1;
    dim3 grid(R / 16, O / (NT * 64));
    dim3 blk(128);
    switch (NT) {
      case 1: kan_wmma<1><<<grid, blk, 0, stream>>>(in, Wf, bias, out, I, O,
                  iRS, iS0, iS1, iS2, oRS, oOB, oQS, oOS); break;
      case 2: kan_wmma<2><<<grid, blk, 0, stream>>>(in, Wf, bias, out, I, O,
                  iRS, iS0, iS1, iS2, oRS, oOB, oQS, oOS); break;
      default: kan_wmma<4><<<grid, blk, 0, stream>>>(in, Wf, bias, out, I, O,
                  iRS, iS0, iS1, iS2, oRS, oOB, oQS, oOS); break;
    }
  };

  // pre: rows r=b*256+d, x2[r,i]=x[b,i,d]; out h[b,o,d]
  kan(x, wbPre, b_pre, hA, 1024, 196, 64,
      /*in*/ 256, 196 * 256, 1, 256, /*out*/ 256, 64 * 256, 1, 256);

  float* hcur = hA; float* hnext = hB;
  const _Float16* wbAmp[3] = {wbAmp0, wbAmp1, wbAmp2};
  const _Float16* wbMix[3] = {wbMix0, wbMix1, wbMix2};
  const float*    bMix[3]  = {b_mix0, b_mix1, b_mix2};
  int C = 64;
  for (int l = 0; l < 3; ++l) {
    int R1 = 4 * C;
    // a = kan(ln(h), w_amp[l])   rows = B*C, I=256, O=512, contiguous
    ln_rows<<<R1 / 8, 256, 0, stream>>>(hcur, hn, 256);
    kan(hn, wbAmp[l], b_amp + (size_t)l * 512, ab, R1, 256, 512,
        /*in*/ R1, 0, 256, 1, /*out*/ R1, 0, 512, 1);
    // an = ln(a) over last dim 512
    ln_rows<<<R1 / 8, 256, 0, stream>>>(ab, an, 512);
    // m = kan(swap(an), w_mix[l]); fused swap+reshape epilogue -> next h
    float* mout = (l == 2) ? (float*)d_out : hnext;
    kan(an, wbMix[l], bMix[l], mout, 2048, C, C,
        /*in*/ 512, C * 512, 1, 512, /*out*/ 512, C * 512, 1, 512);
    float* tmp = hcur; hcur = hnext; hnext = tmp;
    C *= 2;
  }
}
